// FeatureAdaption_26259430048583
// MI455X (gfx1250) — matrix operand order repaired and checked by reading, not compiled, not run on hardware
//
#include <hip/hip_runtime.h>
#include <hip/hip_bf16.h>

typedef __attribute__((ext_vector_type(16))) _Float16 v16h;
typedef __attribute__((ext_vector_type(8)))  _Float16 v8h;
typedef __attribute__((ext_vector_type(8)))  float    v8f;

#define Bn   2
#define CH   256
#define Hh   128
#define Ww   128
#define HW   (Hh*Ww)
#define KKk  9
#define KTOT (CH*KKk)          // 2304
#define KB32 (KTOT/32)         // 72 K-blocks of 32
#define NTIL 16                // N tiles of 16 (N=256)
#define MT   32                // pixels per block (2 M-subtiles of 16)
#define ALDS_STRIDE 264        // halves per A row (256 + 8 pad) -> 528B = 33*16 (b128 aligned)
#define DLDS_TSTRIDE 272       // floats per staged D tile (16 rows * 17)

#define BPK_HALFS (KB32*NTIL*32*16)   // 589824 packed f16 weights (1.18 MB)

// ---------------- Kernel 0: repack weights into WMMA B-fragment order -------
// B-fragment (16-bit, 32x16): lane l holds n = l%16, K = (l/16)*8 + slot (+8 if slot >= 8),
// 16 slots contiguous per lane -> two aligned global_load_b128 per fragment.
__global__ void repack_w(const float* __restrict__ w, _Float16* __restrict__ bpk) {
    int e = blockIdx.x * 256 + threadIdx.x;          // 2304 blocks * 256 = exact
    int slot =  e        & 15;
    int lane = (e >> 4)  & 31;
    int nt   = (e >> 9)  & 15;
    int kbg  =  e >> 13;                             // 0..71
    int o    = nt * 16 + (lane & 15);                // output channel
    int hk   = lane >> 4;
    int kg   = kbg * 32 + (slot < 8 ? hk * 8 + slot : 16 + hk * 8 + (slot - 8));  // global K index (kpos*256 + c)
    int kpos = kg >> 8;
    int c    = kg & 255;
    bpk[e] = (_Float16)w[(o * CH + c) * KKk + kpos];
}

// ---------------- Kernel 1: 1x1 offset conv (256 -> 27) + sigmoid(mask) -----
__global__ void off_conv(const float* __restrict__ off, const float* __restrict__ wof,
                         const float* __restrict__ boff, float* __restrict__ obuf) {
    int p   = blockIdx.x * 256 + threadIdx.x;        // one pixel per thread
    int b   = p / HW;
    int pix = p - b * HW;
    float acc[27];
    #pragma unroll
    for (int oc = 0; oc < 27; ++oc) acc[oc] = boff[oc];
    const float* offp = off + (size_t)b * CH * HW + pix;
    for (int c = 0; c < CH; ++c) {
        float v = offp[(size_t)c * HW];              // coalesced across lanes
        #pragma unroll
        for (int oc = 0; oc < 27; ++oc) acc[oc] += wof[oc * CH + c] * v;  // uniform s_loads
    }
    #pragma unroll
    for (int oc = 18; oc < 27; ++oc) acc[oc] = 1.0f / (1.0f + __expf(-acc[oc]));
    float* op = obuf + (size_t)b * 27 * HW + pix;
    #pragma unroll
    for (int oc = 0; oc < 27; ++oc) op[(size_t)oc * HW] = acc[oc];
}

// ---------------- Kernel 2: implicit-GEMM DCN with f16 WMMA ------------------
// Block: 32 consecutive pixels (one row) x all 256 out channels.
// 8 waves; wave wv owns N-tiles {2wv, 2wv+1} x both 16-pixel M-subtiles.
__global__ void __launch_bounds__(256)
dcn_wmma(const float* __restrict__ xg, const float* __restrict__ obuf,
         const _Float16* __restrict__ bpk, const float* __restrict__ bias,
         float* __restrict__ out)
{
    __shared__ _Float16 Alds[MT * ALDS_STRIDE];      // 32 pixels x 256 ch (f16, padded) ~16.9KB
    __shared__ float    Dlds[NTIL * DLDS_TSTRIDE];   // 16 tiles of 16x16 f32 (padded) ~17.4KB

    const int t    = threadIdx.x;
    const int lane = t & 31;
    const int wv   = t >> 5;                         // wave 0..7; also sampling channel-slab id

    const int base = blockIdx.x * MT;                // 32 consecutive pixels, same row
    const int b    = base / HW;
    const int rem  = base - b * HW;
    const int y    = rem >> 7;                       // /W
    const int x0   = rem & 127;                      // multiple of 32

    // sampling duty: pixel = lane (32 adjacent pixels), channels wv*32 .. wv*32+31
    const float* xb = xg   + ((size_t)b * CH + wv * 32) * HW;
    const float* ob = obuf + (size_t)b * 27 * HW + (y * Ww + x0 + lane);

    v8f c00 = {}, c01 = {}, c10 = {}, c11 = {};      // [M-subtile][N-tile]
    const int nt0 = wv * 2, nt1 = wv * 2 + 1;

    for (int kpos = 0; kpos < KKk; ++kpos) {
        // ---- bilinear setup for pixel `lane`, tap kpos (branchless borders)
        float dy = ob[(size_t)(2 * kpos)     * HW];
        float dx = ob[(size_t)(2 * kpos + 1) * HW];
        float mk = ob[(size_t)(18 + kpos)    * HW];  // already sigmoided
        float ys = (float)(y         + (kpos / 3) - 1) + dy;
        float xs = (float)(x0 + lane + (kpos % 3) - 1) + dx;
        float fy0 = floorf(ys), fx0 = floorf(xs);
        float wy1 = ys - fy0, wx1 = xs - fx0;
        float wy0 = 1.f - wy1, wx0 = 1.f - wx1;
        float vy0 = (fy0 >=  0.f && fy0 <= 127.f) ? 1.f : 0.f;
        float vy1 = (fy0 >= -1.f && fy0 <= 126.f) ? 1.f : 0.f;
        float vx0 = (fx0 >=  0.f && fx0 <= 127.f) ? 1.f : 0.f;
        float vx1 = (fx0 >= -1.f && fx0 <= 126.f) ? 1.f : 0.f;
        int iy0 = min(max((int)fy0,     0), 127);
        int iy1 = min(max((int)fy0 + 1, 0), 127);
        int ix0 = min(max((int)fx0,     0), 127);
        int ix1 = min(max((int)fx0 + 1, 0), 127);
        float w00 = mk * wy0 * wx0 * vy0 * vx0;
        float w01 = mk * wy0 * wx1 * vy0 * vx1;
        float w10 = mk * wy1 * wx0 * vy1 * vx0;
        float w11 = mk * wy1 * wx1 * vy1 * vx1;
        int i00 = iy0 * Ww + ix0, i01 = iy0 * Ww + ix1;
        int i10 = iy1 * Ww + ix0, i11 = iy1 * Ww + ix1;

        __syncthreads();                             // prev tap's LDS readers done
        unsigned* arow32 = (unsigned*)(Alds + lane * ALDS_STRIDE) + wv * 16;
        #pragma unroll
        for (int i = 0; i < 16; ++i) {               // 2 channels per iter, packed b32
            const float* p0 = xb + (size_t)(2 * i) * HW;
            const float* p1 = p0 + HW;
            float v0 = w00 * p0[i00] + w01 * p0[i01] + w10 * p0[i10] + w11 * p0[i11];
            float v1 = w00 * p1[i00] + w01 * p1[i01] + w10 * p1[i10] + w11 * p1[i11];
            unsigned pr = (unsigned)__builtin_bit_cast(unsigned short, (_Float16)v0)
                        | ((unsigned)__builtin_bit_cast(unsigned short, (_Float16)v1) << 16);
            arow32[i] = pr;                          // 32 adjacent-pixel lanes: 2-way bank max
        }
        __syncthreads();

        // ---- 8 K-steps; 4 WMMAs each: 2 A-frags (LDS) x 2 B-frags (L2)
        const int hk = lane >> 4;
        const _Float16* ar0 = Alds + ((lane & 15)     ) * ALDS_STRIDE + hk * 8;  // M-subtile 0
        const _Float16* ar1 = Alds + ((lane & 15) + 16) * ALDS_STRIDE + hk * 8;  // M-subtile 1
        #pragma unroll
        for (int kb = 0; kb < 8; ++kb) {
            v8h lo0 = *(const v8h*)(ar0 + kb * 32);
            v8h hi0 = *(const v8h*)(ar0 + kb * 32 + 16);
            v8h lo1 = *(const v8h*)(ar1 + kb * 32);
            v8h hi1 = *(const v8h*)(ar1 + kb * 32 + 16);
            v16h a0, a1;
            #pragma unroll
            for (int s = 0; s < 8; ++s) {
                a0[s] = lo0[s]; a0[8 + s] = hi0[s];
                a1[s] = lo1[s]; a1[8 + s] = hi1[s];
            }
            const int kbg = kpos * 8 + kb;
            v16h b0 = *(const v16h*)(bpk + (size_t)((kbg * NTIL + nt0) * 32 + lane) * 16);
            v16h b1 = *(const v16h*)(bpk + (size_t)((kbg * NTIL + nt1) * 32 + lane) * 16);
            c00 = __builtin_amdgcn_wmma_f32_16x16x32_f16(false, a0, false, b0, (short)0, c00, false, false);
            c01 = __builtin_amdgcn_wmma_f32_16x16x32_f16(false, a0, false, b1, (short)0, c01, false, false);
            c10 = __builtin_amdgcn_wmma_f32_16x16x32_f16(false, a1, false, b0, (short)0, c10, false, false);
            c11 = __builtin_amdgcn_wmma_f32_16x16x32_f16(false, a1, false, b1, (short)0, c11, false, false);
        }
    }

    // ---- epilogue: bias + ReLU, transpose via LDS, 2 chunks of 16 pixels ----
    const int n  = lane & 15;
    const int mb = (lane >> 4) * 8;
    const float b0v = bias[nt0 * 16 + n];
    const float b1v = bias[nt1 * 16 + n];
    #pragma unroll
    for (int ms = 0; ms < 2; ++ms) {
        if (ms) __syncthreads();                     // chunk-0 stores done before overwrite
        const v8f& d0 = ms ? c10 : c00;
        const v8f& d1 = ms ? c11 : c01;
        #pragma unroll
        for (int r = 0; r < 8; ++r) {
            Dlds[nt0 * DLDS_TSTRIDE + (mb + r) * 17 + n] = fmaxf(d0[r] + b0v, 0.f);
            Dlds[nt1 * DLDS_TSTRIDE + (mb + r) * 17 + n] = fmaxf(d1[r] + b1v, 0.f);
        }
        __syncthreads();
        const int oc = t;                            // one out-channel per thread
        const int nt = oc >> 4, nn = oc & 15;
        float* op = out + ((size_t)b * CH + oc) * HW + y * Ww + x0 + ms * 16;
        #pragma unroll
        for (int mq = 0; mq < 4; ++mq) {
            float4 v;
            v.x = Dlds[nt * DLDS_TSTRIDE + (4 * mq + 0) * 17 + nn];
            v.y = Dlds[nt * DLDS_TSTRIDE + (4 * mq + 1) * 17 + nn];
            v.z = Dlds[nt * DLDS_TSTRIDE + (4 * mq + 2) * 17 + nn];
            v.w = Dlds[nt * DLDS_TSTRIDE + (4 * mq + 3) * 17 + nn];
            *(float4*)(op + 4 * mq) = v;             // 16B-aligned (x0 % 32 == 0)
        }
    }
}

// ---------------------------------------------------------------------------
extern "C" void kernel_launch(void* const* d_in, const int* in_sizes, int n_in,
                              void* d_out, int out_size, void* d_ws, size_t ws_size,
                              hipStream_t stream) {
    const float* x    = (const float*)d_in[0];   // (2,256,128,128)
    const float* off  = (const float*)d_in[1];   // (2,256,128,128)
    const float* wof  = (const float*)d_in[2];   // (27,256,1,1)
    const float* boff = (const float*)d_in[3];   // (27,)
    const float* w    = (const float*)d_in[4];   // (256,256,3,3)
    const float* bias = (const float*)d_in[5];   // (256,)
    float* out = (float*)d_out;

    _Float16* bpk  = (_Float16*)d_ws;                                 // 1.18 MB
    float*    obuf = (float*)((char*)d_ws + (size_t)BPK_HALFS * 2);   // 3.54 MB

    repack_w<<<BPK_HALFS / 256, 256, 0, stream>>>(w, bpk);               // 2304 blocks
    off_conv<<<(Bn * HW) / 256, 256, 0, stream>>>(off, wof, boff, obuf); // 128 blocks
    dcn_wmma<<<(Bn * HW) / MT, 256, 0, stream>>>(x, obuf, bpk, bias, out); // 1024 blocks
}
